// Seq2SeqSumm_81312320848382
// MI455X (gfx1250) — matrix operand order repaired and checked by reading, not compiled, not run on hardware
//
#include <hip/hip_runtime.h>
#include <math.h>

// Problem sizes (fixed by the reference)
#define VSZ  50000
#define EDIM 512
#define HDIM 512
#define BATCH 16
#define TSRC 512
#define TDEC 100
#define GDIM 2048   // 4H

// NOTE: workspace use ~280 MB (emb bf16 51MB, xproj 134MB, attention 17MB, ...).

typedef __bf16 bf16;
typedef __attribute__((ext_vector_type(16))) __bf16 v16bf;
typedef __attribute__((ext_vector_type(8)))  float  v8f;

// ---------------------------------------------------------------------------
// WMMA fragment loaders (16x16x32 bf16; A/B share a contiguous k order, not the ISA table order)
// A (16xK tile, row-major, stride lda): lane l: row = l&15,
//   elems 0..15 -> k = kbase + (l>>4)*16 + e (16 contiguous)
// B (Kx16 operand taken from Bt stored [N,K] row-major, stride ldb):
//   lane l: col = l&15, k = kbase + (l>>4)*16 + e (16 contiguous)
// C/D v8f: elem i -> row = i + (l>>4)*8, col = l&15
// ---------------------------------------------------------------------------
__device__ __forceinline__ v16bf load_A_frag(const bf16* A, int lda, int row0,
                                             int kbase, int lane) {
  int m = row0 + (lane & 15);
  const bf16* p0 = A + (size_t)m * lda + kbase + ((lane >> 4) << 4);
  const bf16* p1 = p0 + 8;
  v16bf r;
#pragma unroll
  for (int e = 0; e < 8; ++e) { r[e] = p0[e]; r[e + 8] = p1[e]; }
  return r;
}

__device__ __forceinline__ v16bf load_B_frag(const bf16* Bt, int ldb, int col0,
                                             int kbase, int lane) {
  const bf16* p = Bt + (size_t)(col0 + (lane & 15)) * ldb + kbase + ((lane >> 4) << 4);
  v16bf r;
#pragma unroll
  for (int e = 0; e < 16; ++e) r[e] = p[e];
  return r;
}

__device__ __forceinline__ v8f wmma_bf16(v16bf a, v16bf b, v8f c) {
  return __builtin_amdgcn_wmma_f32_16x16x32_bf16(false, a, false, b, (short)0, c,
                                                 false, false);
}

__device__ __forceinline__ float sigm(float x) { return 1.0f / (1.0f + __expf(-x)); }

// ---------------------------------------------------------------------------
// Precision prep
// ---------------------------------------------------------------------------
__global__ void k_convert(bf16* dst, const float* src, size_t n) {
  size_t stride = (size_t)gridDim.x * blockDim.x;
  for (size_t i = (size_t)blockIdx.x * blockDim.x + threadIdx.x; i < n; i += stride)
    dst[i] = (bf16)src[i];
}

// dst[c][r] = src[r][c]  (f32 [rows][cols] -> bf16 [cols][rows])
__global__ void k_transpose(bf16* dst, const float* src, int rows, int cols) {
  size_t n = (size_t)rows * cols;
  size_t stride = (size_t)gridDim.x * blockDim.x;
  for (size_t i = (size_t)blockIdx.x * blockDim.x + threadIdx.x; i < n; i += stride) {
    int r = (int)(i / cols), c = (int)(i % cols);
    dst[(size_t)c * rows + r] = (bf16)src[i];
  }
}

// xs[(t*B+b)][e] = emb_bf[ idx[b][t] ][e]   (idx shape [B][T])
__global__ void k_gather(bf16* dst, const bf16* embb, const int* idx, int T) {
  int r = blockIdx.x;                 // r = t*B + b
  int t = r / BATCH, b = r % BATCH;
  int tok = idx[b * T + t];
  const bf16* s = embb + (size_t)tok * EDIM;
  bf16* d = dst + (size_t)r * EDIM;
  for (int e = threadIdx.x; e < EDIM; e += blockDim.x) d[e] = s[e];
}

// ---------------------------------------------------------------------------
// Generic C[M,N] = A[MxK] * Bt[NxK]^T (+bias[N]); one 16x16 tile per wave.
// ---------------------------------------------------------------------------
__global__ __launch_bounds__(256) void k_gemm(const bf16* A, const bf16* Bt,
                                              const float* bias, float* C,
                                              int M, int N, int K,
                                              int lda, int ldb, int ldc) {
  int lane = threadIdx.x & 31;
  int wave = threadIdx.x >> 5;
  int tn_count = N >> 4;
  int tile = blockIdx.x * 8 + wave;
  if (tile >= (M >> 4) * tn_count) return;   // wave-uniform: EXEC stays all-1
  int tm = (tile / tn_count) << 4;
  int tn = (tile % tn_count) << 4;
  v8f acc = {};
  for (int k = 0; k < K; k += 32)
    acc = wmma_bf16(load_A_frag(A, lda, tm, k, lane),
                    load_B_frag(Bt, ldb, tn, k, lane), acc);
  int n = tn + (lane & 15);
  float bb = bias ? bias[n] : 0.0f;
#pragma unroll
  for (int i = 0; i < 8; ++i)
    C[(size_t)(tm + i + ((lane >> 4) << 3)) * ldc + n] = acc[i] + bb;
}

// logits[b][t][v] = dec_out[16xE] @ emb_bf[VxE]^T  (emb stays L2-resident)
__global__ __launch_bounds__(256) void k_logits(const bf16* Adec, const bf16* embb,
                                                float* out, int t) {
  int lane = threadIdx.x & 31;
  int wave = threadIdx.x >> 5;
  int tile = blockIdx.x * 8 + wave;
  if (tile >= (VSZ >> 4)) return;
  int tn = tile << 4;
  v8f acc = {};
  const bf16* brow = embb + (size_t)(tn + (lane & 15)) * EDIM;
  for (int k = 0; k < EDIM; k += 32) {
    __builtin_prefetch(brow + k + 256, 0, 0);      // global_prefetch_b8
    acc = wmma_bf16(load_A_frag(Adec, EDIM, 0, k, lane),
                    load_B_frag(embb, EDIM, tn, k, lane), acc);
  }
  int v = tn + (lane & 15);
#pragma unroll
  for (int i = 0; i < 8; ++i) {
    int b = i + ((lane >> 4) << 3);
    out[((size_t)b * TDEC + t) * VSZ + v] = acc[i];
  }
}

// ---------------------------------------------------------------------------
// Persistent bi-LSTM encoder scan: one workgroup per direction (grid=2).
// h lives in LDS (bf16); c lives in registers with the WMMA C-tile layout.
// Wave w owns gate columns h in [16w, 16w+16) of all four i/f/g/o blocks,
// so the elementwise cell update needs no LDS gate staging.
// ---------------------------------------------------------------------------
__global__ __launch_bounds__(1024) void k_encoder_scan(
    const bf16* WhhF, const bf16* WhhB,              // [2048][512] bf16 [N,K]
    const float* xprojF, const float* xprojB,        // [TS][B][4H] f32 (incl bias)
    const float* init_h, const float* init_c,        // [2][H] f32
    const int* art_lens,
    bf16* enc,                                       // [TS][B][2H] bf16
    float* hfin, float* cfin) {                      // [2][B][H] f32
  __shared__ bf16 h_lds[BATCH][HDIM];
  __shared__ int lens_s[BATCH];
  int dir = blockIdx.x;
  const bf16* Whh = dir ? WhhB : WhhF;
  const float* xproj = dir ? xprojB : xprojF;
  int lane = threadIdx.x & 31;
  int wave = threadIdx.x >> 5;
  int hcol0 = wave << 4;
  if (threadIdx.x < BATCH) {
    int l = art_lens[threadIdx.x];
    lens_s[threadIdx.x] = l < 1 ? 1 : l;
  }
  for (int i = threadIdx.x; i < BATCH * HDIM; i += 1024)
    h_lds[i >> 9][i & 511] = (bf16)init_h[dir * HDIM + (i & 511)];
  v8f c = {};
  {
    float cv = init_c[dir * HDIM + hcol0 + (lane & 15)];
#pragma unroll
    for (int i = 0; i < 8; ++i) c[i] = cv;
  }
  __syncthreads();
  for (int s = 0; s < TSRC; ++s) {
    int t = dir ? (TSRC - 1 - s) : s;
    v8f a0 = {}, a1 = {}, a2 = {}, a3 = {};
    for (int k = 0; k < HDIM; k += 32) {
      v16bf a = load_A_frag(&h_lds[0][0], HDIM, 0, k, lane);
      a0 = wmma_bf16(a, load_B_frag(Whh, HDIM, 0 * HDIM + hcol0, k, lane), a0);
      a1 = wmma_bf16(a, load_B_frag(Whh, HDIM, 1 * HDIM + hcol0, k, lane), a1);
      a2 = wmma_bf16(a, load_B_frag(Whh, HDIM, 2 * HDIM + hcol0, k, lane), a2);
      a3 = wmma_bf16(a, load_B_frag(Whh, HDIM, 3 * HDIM + hcol0, k, lane), a3);
    }
    __syncthreads();               // everyone is done reading old h
    int h = hcol0 + (lane & 15);
    const float* xp = xproj + (size_t)t * BATCH * GDIM;
#pragma unroll
    for (int i = 0; i < 8; ++i) {
      int b = i + ((lane >> 4) << 3);
      const float* xpb = xp + (size_t)b * GDIM;
      float gi = a0[i] + xpb[h];
      float gf = a1[i] + xpb[HDIM + h];
      float gg = a2[i] + xpb[2 * HDIM + h];
      float go = a3[i] + xpb[3 * HDIM + h];
      float cn = sigm(gf) * c[i] + sigm(gi) * tanhf(gg);
      float hn = sigm(go) * tanhf(cn);
      bool valid = t < lens_s[b];  // packed-sequence: freeze state on pads
      float hv = valid ? hn : (float)h_lds[b][h];
      if (valid) c[i] = cn;
      h_lds[b][h] = (bf16)hv;
      enc[((size_t)t * BATCH + b) * (2 * HDIM) + dir * HDIM + h] = (bf16)hv;
    }
    __syncthreads();
  }
#pragma unroll
  for (int i = 0; i < 8; ++i) {
    int b = i + ((lane >> 4) << 3);
    int h = hcol0 + (lane & 15);
    hfin[((size_t)dir * BATCH + b) * HDIM + h] = (float)h_lds[b][h];
    cfin[((size_t)dir * BATCH + b) * HDIM + h] = c[i];
  }
}

// ---------------------------------------------------------------------------
// Decoder init: init_h/c = [hf|hb]@dec_{h,c}_w^T, mean_ctx, init_out = proj(...)
// ---------------------------------------------------------------------------
__global__ __launch_bounds__(1024) void k_init_dec(
    const float* hfin, const float* cfin,
    const bf16* dhw, const bf16* dcw,          // [H][2H] bf16 [N,K]
    const float* attn, const int* art_lens,
    const bf16* pw1, const float* pb1, const bf16* pw2,
    bf16* dh_bf, float* dc, bf16* out0) {
  extern __shared__ char smem[];
  bf16* A_lds = (bf16*)smem;                   // [16][1024]
  bf16* x_lds = (bf16*)(smem + 32768);         // [16][512]  init_h
  bf16* y_lds = (bf16*)(smem + 49152);         // [16][512]  mean_ctx
  bf16* p_lds = (bf16*)(smem + 65536);         // [16][512]  tanh proj
  __shared__ int lens_s[BATCH];
  int lane = threadIdx.x & 31, wave = threadIdx.x >> 5;
  int n0 = wave << 4;
  if (threadIdx.x < BATCH) {
    int l = art_lens[threadIdx.x];
    lens_s[threadIdx.x] = l < 1 ? 1 : l;
  }
  for (int i = threadIdx.x; i < BATCH * 2 * HDIM; i += 1024) {
    int b = i >> 10, k = i & 1023;
    float v = (k < HDIM) ? hfin[(size_t)b * HDIM + k]
                         : hfin[(size_t)(BATCH + b) * HDIM + (k - HDIM)];
    A_lds[i] = (bf16)v;
  }
  __syncthreads();
  {
    v8f acc = {};
    for (int k = 0; k < 2 * HDIM; k += 32)
      acc = wmma_bf16(load_A_frag(A_lds, 2 * HDIM, 0, k, lane),
                      load_B_frag(dhw, 2 * HDIM, n0, k, lane), acc);
    int n = n0 + (lane & 15);
#pragma unroll
    for (int i = 0; i < 8; ++i) {
      int b = i + ((lane >> 4) << 3);
      bf16 hv = (bf16)acc[i];
      x_lds[b * HDIM + n] = hv;
      dh_bf[(size_t)b * HDIM + n] = hv;
    }
  }
  __syncthreads();
  for (int i = threadIdx.x; i < BATCH * 2 * HDIM; i += 1024) {
    int b = i >> 10, k = i & 1023;
    float v = (k < HDIM) ? cfin[(size_t)b * HDIM + k]
                         : cfin[(size_t)(BATCH + b) * HDIM + (k - HDIM)];
    A_lds[i] = (bf16)v;
  }
  __syncthreads();
  {
    v8f acc = {};
    for (int k = 0; k < 2 * HDIM; k += 32)
      acc = wmma_bf16(load_A_frag(A_lds, 2 * HDIM, 0, k, lane),
                      load_B_frag(dcw, 2 * HDIM, n0, k, lane), acc);
    int n = n0 + (lane & 15);
#pragma unroll
    for (int i = 0; i < 8; ++i)
      dc[(size_t)(i + ((lane >> 4) << 3)) * HDIM + n] = acc[i];
  }
  // mean context over valid timesteps
  for (int i = threadIdx.x; i < BATCH * HDIM; i += 1024) {
    int b = i >> 9, h = i & 511;
    int L = lens_s[b];
    float s = 0.0f;
    for (int t = 0; t < L; ++t) s += attn[((size_t)t * BATCH + b) * HDIM + h];
    y_lds[i] = (bf16)(s / (float)L);
  }
  __syncthreads();
  {
    v8f acc = {};
    for (int k = 0; k < HDIM; k += 32) {
      acc = wmma_bf16(load_A_frag(x_lds, HDIM, 0, k, lane),
                      load_B_frag(pw1, 2 * HDIM, n0, k, lane), acc);
      acc = wmma_bf16(load_A_frag(y_lds, HDIM, 0, k, lane),
                      load_B_frag(pw1 + HDIM, 2 * HDIM, n0, k, lane), acc);
    }
    int n = n0 + (lane & 15);
    float bb = pb1[n];
#pragma unroll
    for (int i = 0; i < 8; ++i)
      p_lds[(i + ((lane >> 4) << 3)) * HDIM + n] = (bf16)tanhf(acc[i] + bb);
  }
  __syncthreads();
  {
    v8f acc = {};
    for (int k = 0; k < HDIM; k += 32)
      acc = wmma_bf16(load_A_frag(p_lds, HDIM, 0, k, lane),
                      load_B_frag(pw2, HDIM, n0, k, lane), acc);
    int n = n0 + (lane & 15);
#pragma unroll
    for (int i = 0; i < 8; ++i)
      out0[(size_t)(i + ((lane >> 4) << 3)) * EDIM + n] = (bf16)acc[i];
  }
}

// ---------------------------------------------------------------------------
// One decoder step: LSTM cell + attention + projection (single WGP, 1024 thr)
// ---------------------------------------------------------------------------
__global__ __launch_bounds__(1024) void k_dec_cell(
    const bf16* dWhh,            // [2048][512]
    const bf16* dWihR,           // dWih + E (out half), ldb = 2E
    const float* tgtproj,        // [TD][B][4H] f32 (e_t proj + bias)
    const bf16* wqT,             // [512][512]  attn_wq^T
    const float* attn, const int* art_lens,
    const bf16* pw1, const float* pb1, const bf16* pw2,
    bf16* dh_bf, float* dc, const bf16* out_prev, bf16* out_next, int t) {
  extern __shared__ char smem[];
  bf16*  h_lds = (bf16*)smem;                      // 16x512 bf16
  bf16*  o_lds = (bf16*)(smem + 16384);            // 16x512 bf16
  float* q_lds = (float*)(smem + 32768);           // 16x512 f32
  float* s_lds = (float*)(smem + 65536);           // 16x512 f32 (scores)
  bf16*  c_lds = (bf16*)(smem + 98304);            // ctx 16x512 bf16
  bf16*  p_lds = (bf16*)(smem + 114688);           // proj 16x512 bf16
  float* red   = (float*)(smem + 131072);          // 32 floats
  __shared__ int lens_s[BATCH];
  int lane = threadIdx.x & 31, wave = threadIdx.x >> 5;
  int hcol0 = wave << 4;
  if (threadIdx.x < BATCH) {
    int l = art_lens[threadIdx.x];
    lens_s[threadIdx.x] = l < 1 ? 1 : l;
  }
  for (int i = threadIdx.x; i < BATCH * HDIM; i += 1024) {
    h_lds[i] = dh_bf[i];
    o_lds[i] = out_prev[i];
  }
  __syncthreads();
  // gates = tgtproj[t] + out@dWihR^T + h@dWhh^T
  v8f a0 = {}, a1 = {}, a2 = {}, a3 = {};
  for (int k = 0; k < HDIM; k += 32) {
    v16bf a = load_A_frag(h_lds, HDIM, 0, k, lane);
    a0 = wmma_bf16(a, load_B_frag(dWhh, HDIM, 0 * HDIM + hcol0, k, lane), a0);
    a1 = wmma_bf16(a, load_B_frag(dWhh, HDIM, 1 * HDIM + hcol0, k, lane), a1);
    a2 = wmma_bf16(a, load_B_frag(dWhh, HDIM, 2 * HDIM + hcol0, k, lane), a2);
    a3 = wmma_bf16(a, load_B_frag(dWhh, HDIM, 3 * HDIM + hcol0, k, lane), a3);
  }
  for (int k = 0; k < EDIM; k += 32) {
    v16bf a = load_A_frag(o_lds, EDIM, 0, k, lane);
    a0 = wmma_bf16(a, load_B_frag(dWihR, 2 * EDIM, 0 * HDIM + hcol0, k, lane), a0);
    a1 = wmma_bf16(a, load_B_frag(dWihR, 2 * EDIM, 1 * HDIM + hcol0, k, lane), a1);
    a2 = wmma_bf16(a, load_B_frag(dWihR, 2 * EDIM, 2 * HDIM + hcol0, k, lane), a2);
    a3 = wmma_bf16(a, load_B_frag(dWihR, 2 * EDIM, 3 * HDIM + hcol0, k, lane), a3);
  }
  __syncthreads();
  {
    int h = hcol0 + (lane & 15);
    const float* xp = tgtproj + (size_t)t * BATCH * GDIM;
#pragma unroll
    for (int i = 0; i < 8; ++i) {
      int b = i + ((lane >> 4) << 3);
      const float* xpb = xp + (size_t)b * GDIM;
      float gi = a0[i] + xpb[h];
      float gf = a1[i] + xpb[HDIM + h];
      float gg = a2[i] + xpb[2 * HDIM + h];
      float go = a3[i] + xpb[3 * HDIM + h];
      float cp = dc[(size_t)b * HDIM + h];
      float cn = sigm(gf) * cp + sigm(gi) * tanhf(gg);
      float hn = sigm(go) * tanhf(cn);
      dc[(size_t)b * HDIM + h] = cn;
      bf16 hb = (bf16)hn;
      h_lds[b * HDIM + h] = hb;
      dh_bf[(size_t)b * HDIM + h] = hb;
    }
  }
  __syncthreads();
  // q = h @ attn_wq
  {
    v8f acc = {};
    for (int k = 0; k < HDIM; k += 32)
      acc = wmma_bf16(load_A_frag(h_lds, HDIM, 0, k, lane),
                      load_B_frag(wqT, HDIM, hcol0, k, lane), acc);
    int n = hcol0 + (lane & 15);
#pragma unroll
    for (int i = 0; i < 8; ++i)
      q_lds[(i + ((lane >> 4) << 3)) * HDIM + n] = acc[i];
  }
  __syncthreads();
  // masked scores
  for (int i = threadIdx.x; i < BATCH * TSRC; i += 1024) {
    int b = i >> 9, ts = i & 511;
    float s = -3.0e38f;
    if (ts < lens_s[b]) {
      const float* ap = attn + ((size_t)ts * BATCH + b) * HDIM;
      const float* qp = q_lds + b * HDIM;
      float acc = 0.0f;
      for (int h = 0; h < HDIM; ++h) acc += qp[h] * ap[h];
      s = acc;
    }
    s_lds[b * TSRC + ts] = s;
  }
  __syncthreads();
  if (threadIdx.x < BATCH) {
    const float* sp = s_lds + threadIdx.x * TSRC;
    float m = sp[0];
    for (int ts = 1; ts < TSRC; ++ts) m = fmaxf(m, sp[ts]);
    red[threadIdx.x] = m;
  }
  __syncthreads();
  for (int i = threadIdx.x; i < BATCH * TSRC; i += 1024)
    s_lds[i] = __expf(s_lds[i] - red[i >> 9]);
  __syncthreads();
  if (threadIdx.x < BATCH) {
    const float* sp = s_lds + threadIdx.x * TSRC;
    float sum = 0.0f;
    for (int ts = 0; ts < TSRC; ++ts) sum += sp[ts];
    red[BATCH + threadIdx.x] = 1.0f / sum;
  }
  __syncthreads();
  // ctx = softmax(scores) @ attention  (attention tensor is L2-resident)
  for (int i = threadIdx.x; i < BATCH * HDIM; i += 1024) {
    int b = i >> 9, h = i & 511;
    const float* sp = s_lds + b * TSRC;
    float acc = 0.0f;
    for (int ts = 0; ts < TSRC; ++ts)
      acc += sp[ts] * attn[((size_t)ts * BATCH + b) * HDIM + h];
    c_lds[i] = (bf16)(acc * red[BATCH + b]);
  }
  __syncthreads();
  // dec_out = tanh([h|ctx]@pw1^T + b1) @ pw2^T
  {
    v8f acc = {};
    for (int k = 0; k < HDIM; k += 32) {
      acc = wmma_bf16(load_A_frag(h_lds, HDIM, 0, k, lane),
                      load_B_frag(pw1, 2 * HDIM, hcol0, k, lane), acc);
      acc = wmma_bf16(load_A_frag(c_lds, HDIM, 0, k, lane),
                      load_B_frag(pw1 + HDIM, 2 * HDIM, hcol0, k, lane), acc);
    }
    int n = hcol0 + (lane & 15);
    float bb = pb1[n];
#pragma unroll
    for (int i = 0; i < 8; ++i)
      p_lds[(i + ((lane >> 4) << 3)) * HDIM + n] = (bf16)tanhf(acc[i] + bb);
  }
  __syncthreads();
  {
    v8f acc = {};
    for (int k = 0; k < HDIM; k += 32)
      acc = wmma_bf16(load_A_frag(p_lds, HDIM, 0, k, lane),
                      load_B_frag(pw2, HDIM, hcol0, k, lane), acc);
    int n = hcol0 + (lane & 15);
#pragma unroll
    for (int i = 0; i < 8; ++i)
      out_next[(size_t)(i + ((lane >> 4) << 3)) * EDIM + n] = (bf16)acc[i];
  }
}

// ---------------------------------------------------------------------------
extern "C" void kernel_launch(void* const* d_in, const int* in_sizes, int n_in,
                              void* d_out, int out_size, void* d_ws, size_t ws_size,
                              hipStream_t stream) {
  (void)in_sizes; (void)n_in; (void)out_size; (void)ws_size;
  const int*   article    = (const int*)  d_in[0];
  const int*   art_lens   = (const int*)  d_in[1];
  const int*   abstract   = (const int*)  d_in[2];
  const float* emb        = (const float*)d_in[3];
  const float* eWih_f     = (const float*)d_in[4];
  const float* eWhh_f     = (const float*)d_in[5];
  const float* eb_f       = (const float*)d_in[6];
  const float* eWih_b     = (const float*)d_in[7];
  const float* eWhh_b     = (const float*)d_in[8];
  const float* eb_b       = (const float*)d_in[9];
  const float* init_enc_h = (const float*)d_in[10];
  const float* init_enc_c = (const float*)d_in[11];
  const float* dec_h_w    = (const float*)d_in[12];
  const float* dec_c_w    = (const float*)d_in[13];
  const float* attn_wm    = (const float*)d_in[14];
  const float* attn_wq    = (const float*)d_in[15];
  const float* proj_w1    = (const float*)d_in[16];
  const float* proj_b1    = (const float*)d_in[17];
  const float* proj_w2    = (const float*)d_in[18];
  const float* dWih       = (const float*)d_in[19];
  const float* dWhh       = (const float*)d_in[20];
  const float* db         = (const float*)d_in[21];
  float* out = (float*)d_out;

  char* ws = (char*)d_ws;
  size_t off = 0;
  auto alloc = [&](size_t bytes) -> char* {
    char* p = ws + off;
    off = (off + bytes + 255) & ~(size_t)255;
    return p;
  };
  bf16*  emb_bf  = (bf16*) alloc((size_t)VSZ * EDIM * 2);
  bf16*  xs_bf   = (bf16*) alloc((size_t)TSRC * BATCH * EDIM * 2);
  bf16*  tgt_bf  = (bf16*) alloc((size_t)TDEC * BATCH * EDIM * 2);
  bf16*  WihF_bf = (bf16*) alloc((size_t)GDIM * EDIM * 2);
  bf16*  WhhF_bf = (bf16*) alloc((size_t)GDIM * HDIM * 2);
  bf16*  WihB_bf = (bf16*) alloc((size_t)GDIM * EDIM * 2);
  bf16*  WhhB_bf = (bf16*) alloc((size_t)GDIM * HDIM * 2);
  bf16*  dWih_bf = (bf16*) alloc((size_t)GDIM * 2 * EDIM * 2);
  bf16*  dWhh_bf = (bf16*) alloc((size_t)GDIM * HDIM * 2);
  bf16*  dhw_bf  = (bf16*) alloc((size_t)HDIM * 2 * HDIM * 2);
  bf16*  dcw_bf  = (bf16*) alloc((size_t)HDIM * 2 * HDIM * 2);
  bf16*  wmT_bf  = (bf16*) alloc((size_t)HDIM * 2 * HDIM * 2);
  bf16*  wqT_bf  = (bf16*) alloc((size_t)HDIM * HDIM * 2);
  bf16*  pw1_bf  = (bf16*) alloc((size_t)HDIM * 2 * HDIM * 2);
  bf16*  pw2_bf  = (bf16*) alloc((size_t)EDIM * HDIM * 2);
  float* xprojF  = (float*)alloc((size_t)TSRC * BATCH * GDIM * 4);
  float* xprojB  = (float*)alloc((size_t)TSRC * BATCH * GDIM * 4);
  bf16*  enc_bf  = (bf16*) alloc((size_t)TSRC * BATCH * 2 * HDIM * 2);
  float* hfin    = (float*)alloc((size_t)2 * BATCH * HDIM * 4);
  float* cfin    = (float*)alloc((size_t)2 * BATCH * HDIM * 4);
  float* attnTB  = (float*)alloc((size_t)TSRC * BATCH * HDIM * 4);
  float* tgtproj = (float*)alloc((size_t)TDEC * BATCH * GDIM * 4);
  float* dcst    = (float*)alloc((size_t)BATCH * HDIM * 4);
  bf16*  dh_bf   = (bf16*) alloc((size_t)BATCH * HDIM * 2);
  bf16*  outslot = (bf16*) alloc((size_t)(TDEC + 1) * BATCH * EDIM * 2);

  // -- bf16 weight prep (emb stays L2-resident at 51 MB) --
  k_convert<<<4096, 256, 0, stream>>>(emb_bf, emb, (size_t)VSZ * EDIM);
  k_convert<<<256, 256, 0, stream>>>(WihF_bf, eWih_f, (size_t)GDIM * EDIM);
  k_convert<<<256, 256, 0, stream>>>(WhhF_bf, eWhh_f, (size_t)GDIM * HDIM);
  k_convert<<<256, 256, 0, stream>>>(WihB_bf, eWih_b, (size_t)GDIM * EDIM);
  k_convert<<<256, 256, 0, stream>>>(WhhB_bf, eWhh_b, (size_t)GDIM * HDIM);
  k_convert<<<256, 256, 0, stream>>>(dWih_bf, dWih, (size_t)GDIM * 2 * EDIM);
  k_convert<<<256, 256, 0, stream>>>(dWhh_bf, dWhh, (size_t)GDIM * HDIM);
  k_convert<<<256, 256, 0, stream>>>(dhw_bf, dec_h_w, (size_t)HDIM * 2 * HDIM);
  k_convert<<<256, 256, 0, stream>>>(dcw_bf, dec_c_w, (size_t)HDIM * 2 * HDIM);
  k_convert<<<256, 256, 0, stream>>>(pw1_bf, proj_w1, (size_t)HDIM * 2 * HDIM);
  k_convert<<<256, 256, 0, stream>>>(pw2_bf, proj_w2, (size_t)EDIM * HDIM);
  k_transpose<<<256, 256, 0, stream>>>(wmT_bf, attn_wm, 2 * HDIM, HDIM);
  k_transpose<<<256, 256, 0, stream>>>(wqT_bf, attn_wq, HDIM, HDIM);

  // -- embedding gathers --
  k_gather<<<TSRC * BATCH, 256, 0, stream>>>(xs_bf, emb_bf, article, TSRC);
  k_gather<<<TDEC * BATCH, 256, 0, stream>>>(tgt_bf, emb_bf, abstract, TDEC);

  // -- input projections hoisted out of the serial scans --
  k_gemm<<<8192, 256, 0, stream>>>(xs_bf, WihF_bf, eb_f, xprojF,
                                   TSRC * BATCH, GDIM, EDIM, EDIM, EDIM, GDIM);
  k_gemm<<<8192, 256, 0, stream>>>(xs_bf, WihB_bf, eb_b, xprojB,
                                   TSRC * BATCH, GDIM, EDIM, EDIM, EDIM, GDIM);
  k_gemm<<<1600, 256, 0, stream>>>(tgt_bf, dWih_bf, db, tgtproj,
                                   TDEC * BATCH, GDIM, EDIM, EDIM, 2 * EDIM, GDIM);

  // -- bi-LSTM encoder (2 persistent workgroups) --
  k_encoder_scan<<<2, 1024, 0, stream>>>(WhhF_bf, WhhB_bf, xprojF, xprojB,
                                         init_enc_h, init_enc_c, art_lens,
                                         enc_bf, hfin, cfin);

  // -- attention = enc @ attn_wm (rows = t*B+b) --
  k_gemm<<<2048, 256, 0, stream>>>(enc_bf, wmT_bf, nullptr, attnTB,
                                   TSRC * BATCH, HDIM, 2 * HDIM,
                                   2 * HDIM, 2 * HDIM, HDIM);

  // -- decoder init state --
  k_init_dec<<<1, 1024, 81920, stream>>>(hfin, cfin, dhw_bf, dcw_bf, attnTB,
                                         art_lens, pw1_bf, proj_b1, pw2_bf,
                                         dh_bf, dcst, outslot);

  // -- decoder: serial cell (1 WGP) + parallel vocab logits per step --
  for (int t = 0; t < TDEC; ++t) {
    k_dec_cell<<<1, 1024, 131200, stream>>>(
        dWhh_bf, dWih_bf + EDIM, tgtproj, wqT_bf, attnTB, art_lens,
        pw1_bf, proj_b1, pw2_bf, dh_bf, dcst,
        outslot + (size_t)t * BATCH * EDIM,
        outslot + (size_t)(t + 1) * BATCH * EDIM, t);
    k_logits<<<391, 256, 0, stream>>>(outslot + (size_t)(t + 1) * BATCH * EDIM,
                                      emb_bf, out, t);
  }
}
